// LSTMEncoder_73770358276748
// MI455X (gfx1250) — matrix operand order repaired and checked by reading, not compiled, not run on hardware
//
#include <hip/hip_runtime.h>
#include <hip/hip_bf16.h>
#include <cstdint>
#include <cstddef>

typedef __attribute__((ext_vector_type(16))) _Float16 v16h;
typedef __attribute__((ext_vector_type(8)))  _Float16 v8h;
typedef __attribute__((ext_vector_type(8)))  float    v8f;

namespace {
constexpr int NB   = 64;     // batch
constexpr int NT   = 512;    // time
constexpr int NC   = 1024;   // channels == hidden
constexpr int NG   = 4096;   // 4*H
constexpr long NBT = (long)NB * NT;
}

// ---------------- async global->LDS (CDNA5 ASYNCcnt path), guarded ----------------
#if __has_builtin(__builtin_amdgcn_global_load_async_to_lds_b128)
#define USE_ASYNC_LDS 1
#else
#define USE_ASYNC_LDS 0
#endif

typedef __attribute__((vector_size(4 * sizeof(int)))) int v4i;
typedef __attribute__((address_space(1))) v4i* as1_v4i_p;
typedef __attribute__((address_space(3))) v4i* as3_v4i_p;

// copy 16 bytes (8 halves) global -> LDS
__device__ __forceinline__ void copy16_g2l(const _Float16* g, _Float16* l) {
#if USE_ASYNC_LDS
  __builtin_amdgcn_global_load_async_to_lds_b128(
      (as1_v4i_p)(v4i*)(void*)const_cast<_Float16*>(g),
      (as3_v4i_p)(v4i*)(void*)l, 0, 0);
#else
  *(v8h*)l = *(const v8h*)g;
#endif
}

__device__ __forceinline__ void wait_async_copies() {
#if USE_ASYNC_LDS
#if __has_builtin(__builtin_amdgcn_s_wait_asynccnt)
  __builtin_amdgcn_s_wait_asynccnt(0);
#else
  asm volatile("s_wait_asynccnt 0" ::: "memory");
#endif
#endif
}

// ---------------- WMMA fragment helpers (f16, 16x16x32) ----------------
// A (16x32, MxK), row-major source, ld in elements.
// Lanes 0-15: row=lane,  K = {0..7} then {16..23}
// Lanes 16-31: row=lane-16, K = {8..15} then {24..31}
__device__ __forceinline__ v16h frag_a(const _Float16* base, int ld) {
  const int lane = threadIdx.x & 31;
  const _Float16* p = base + (size_t)(lane & 15) * ld + ((lane >> 4) << 3);
  v8h lo = *(const v8h*)p;
  v8h hi = *(const v8h*)(p + 16);
  return __builtin_shufflevector(lo, hi, 0,1,2,3,4,5,6,7,8,9,10,11,12,13,14,15);
}

// B (32x16, KxN) taken from an [N,K] row-major matrix (i.e. W as given).
// Lane n = lane%16 holds column n, K = 8*(lane/16)+{0..7} then 16+8*(lane/16)+{0..7}.
__device__ __forceinline__ v16h frag_b(const _Float16* base, int ld) {
  const int lane = threadIdx.x & 31;
  const _Float16* p = base + (size_t)(lane & 15) * ld + ((lane >> 4) << 3);
  v8h lo = *(const v8h*)p;
  v8h hi = *(const v8h*)(p + 16);
  return __builtin_shufflevector(lo, hi, 0,1,2,3,4,5,6,7,8,9,10,11,12,13,14,15);
}

// ---------------- utility kernels ----------------
__global__ __launch_bounds__(256) void f32_to_f16_kernel(const float* __restrict__ s,
                                                         _Float16* __restrict__ d, int n) {
  int i = blockIdx.x * 256 + threadIdx.x;
  if (i < n) d[i] = (_Float16)s[i];
}

__global__ __launch_bounds__(256) void init_state_kernel(_Float16* hA, _Float16* hB, float* c) {
  int i = blockIdx.x * 256 + threadIdx.x;   // 512*256 == 131072 == 2*64*1024
  hA[i] = (_Float16)0.f;
  hB[i] = (_Float16)0.f;
  c[i]  = 0.f;
}

__device__ __forceinline__ void block_reduce2(float& s, float& q) {
  #pragma unroll
  for (int o = 16; o > 0; o >>= 1) { s += __shfl_xor(s, o); q += __shfl_xor(q, o); }
  __shared__ float rs[8], rq[8];
  const int wave = threadIdx.x >> 5, lane = threadIdx.x & 31;
  if (lane == 0) { rs[wave] = s; rq[wave] = q; }
  __syncthreads();
  s = 0.f; q = 0.f;
  #pragma unroll
  for (int i = 0; i < 8; ++i) { s += rs[i]; q += rq[i]; }
}

// LayerNorm over C=1024, output f16 (feeds the WMMA GEMMs)
__global__ __launch_bounds__(256) void ln_x_kernel(const float* __restrict__ x,
                                                   const float* __restrict__ g,
                                                   const float* __restrict__ b,
                                                   _Float16* __restrict__ out) {
  const long row = blockIdx.x;
  const int tid = threadIdx.x;
  const float* xr = x + row * NC;
  float v[4], s = 0.f, q = 0.f;
  #pragma unroll
  for (int i = 0; i < 4; ++i) { v[i] = xr[tid + i * 256]; s += v[i]; q += v[i] * v[i]; }
  block_reduce2(s, q);
  const float mean = s * (1.f / NC);
  const float inv  = rsqrtf(q * (1.f / NC) - mean * mean + 1e-5f);
  #pragma unroll
  for (int i = 0; i < 4; ++i) {
    const int c = tid + i * 256;
    out[row * NC + c] = (_Float16)((v[i] - mean) * inv * g[c] + b[c]);
  }
}

// Final in-place LayerNorm over 2H=2048 of d_out
__global__ __launch_bounds__(256) void ln_out_kernel(const float* __restrict__ g,
                                                     const float* __restrict__ b,
                                                     float* __restrict__ io) {
  const long row = blockIdx.x;
  const int tid = threadIdx.x;
  float* xr = io + row * (2 * NC);
  float v[8], s = 0.f, q = 0.f;
  #pragma unroll
  for (int i = 0; i < 8; ++i) { v[i] = xr[tid + i * 256]; s += v[i]; q += v[i] * v[i]; }
  block_reduce2(s, q);
  const float mean = s * (1.f / (2 * NC));
  const float inv  = rsqrtf(q * (1.f / (2 * NC)) - mean * mean + 1e-5f);
  #pragma unroll
  for (int i = 0; i < 8; ++i) {
    const int c = tid + i * 256;
    xr[c] = (v[i] - mean) * inv * g[c] + b[c];
  }
}

// ---------------- big GEMM: xg = xn[32768x1024] * Wih^T -> [32768x4096] (f16, +bias) ----
// Block tile 128x128, 8 waves, each wave 4(M)x2(N) fragments, K chunk 32, LDS staged
// via async global->LDS (ASYNCcnt) when available.
__global__ __launch_bounds__(256) void gemm_xproj_kernel(
    const _Float16* __restrict__ xn,
    const _Float16* __restrict__ Wf, const _Float16* __restrict__ Wb,
    const float* __restrict__ bf, const float* __restrict__ bb,
    _Float16* __restrict__ xgf, _Float16* __restrict__ xgb) {
  const int dir = blockIdx.z;
  const _Float16* W   = dir ? Wb : Wf;
  const float*    bia = dir ? bb : bf;
  _Float16*       xg  = dir ? xgb : xgf;

  __shared__ _Float16 As[128 * 32];  // 8 KB
  __shared__ _Float16 Bs[128 * 32];  // 8 KB

  const int m0 = blockIdx.y * 128;
  const int n0 = blockIdx.x * 128;
  const int tid  = threadIdx.x;
  const int lane = tid & 31;
  const int wave = tid >> 5;
  const int mg = wave >> 2;   // 0..1 -> mfrags mg*4..mg*4+3
  const int ng = wave & 3;    // 0..3 -> nfrags ng*2..ng*2+1

  v8f acc[4][2];
  #pragma unroll
  for (int mi = 0; mi < 4; ++mi)
    #pragma unroll
    for (int ni = 0; ni < 2; ++ni)
      #pragma unroll
      for (int r = 0; r < 8; ++r) acc[mi][ni][r] = 0.f;

  for (int k0 = 0; k0 < NC; k0 += 32) {
    // stage A (128x32) and B (128x32) tiles: 512 chunks of 8 halves each
    #pragma unroll
    for (int i = tid; i < 512; i += 256) {
      const int r = i >> 2, sc = (i & 3) << 3;
      copy16_g2l(&xn[(size_t)(m0 + r) * NC + k0 + sc], &As[r * 32 + sc]);
      copy16_g2l(&W [(size_t)(n0 + r) * NC + k0 + sc], &Bs[r * 32 + sc]);
    }
    wait_async_copies();
    __syncthreads();

    v16h af[4];
    #pragma unroll
    for (int mi = 0; mi < 4; ++mi) af[mi] = frag_a(&As[(mg * 4 + mi) * 16 * 32], 32);
    #pragma unroll
    for (int ni = 0; ni < 2; ++ni) {
      v16h bfr = frag_b(&Bs[(ng * 2 + ni) * 16 * 32], 32);
      #pragma unroll
      for (int mi = 0; mi < 4; ++mi)
        acc[mi][ni] = __builtin_amdgcn_wmma_f32_16x16x32_f16(
            false, af[mi], false, bfr, (short)0, acc[mi][ni], false, false);
    }
    __syncthreads();
  }

  // epilogue: + bias, convert to f16
  #pragma unroll
  for (int mi = 0; mi < 4; ++mi)
    #pragma unroll
    for (int ni = 0; ni < 2; ++ni) {
      const int n  = n0 + (ng * 2 + ni) * 16 + (lane & 15);
      const int mo = m0 + (mg * 4 + mi) * 16 + (lane >> 4) * 8;
      const float bv = bia[n];
      #pragma unroll
      for (int r = 0; r < 8; ++r)
        xg[(size_t)(mo + r) * NG + n] = (_Float16)(acc[mi][ni][r] + bv);
    }
}

// ---------------- one LSTM timestep, both directions (blockIdx.z) ----------------
// Block owns all 64 batch rows x one 64-unit strip (=> 64x256 gate columns).
// Weights are L2-resident f16; fragments loaded straight from global.
__global__ __launch_bounds__(256) void lstm_step_kernel(
    const _Float16* __restrict__ xgf, const _Float16* __restrict__ xgb,
    const _Float16* __restrict__ Whf, const _Float16* __restrict__ Whb,
    const _Float16* __restrict__ hprev, _Float16* __restrict__ hnext,
    float* __restrict__ cbuf, float* __restrict__ out, int step) {
  const int dir = blockIdx.z;
  const int t   = dir ? (NT - 1 - step) : step;
  const _Float16* xg = dir ? xgb : xgf;
  const _Float16* Wh = dir ? Whb : Whf;
  const _Float16* hp = hprev + (size_t)dir * NB * NC;
  _Float16*       hn = hnext + (size_t)dir * NB * NC;
  float*          cc = cbuf  + (size_t)dir * NB * NC;

  const int u0   = blockIdx.x * 64;
  const int tid  = threadIdx.x;
  const int lane = tid & 31;
  const int wave = tid >> 5;              // wave owns nfrags {2w, 2w+1} of 16

  __shared__ float gs[64 * 256];          // 64 KB: [b][gate*64 + uloc]

  v8f acc[4][2];
  #pragma unroll
  for (int mi = 0; mi < 4; ++mi)
    #pragma unroll
    for (int ni = 0; ni < 2; ++ni)
      #pragma unroll
      for (int r = 0; r < 8; ++r) acc[mi][ni][r] = 0.f;

  for (int k0 = 0; k0 < NC; k0 += 32) {
    v16h af[4];
    #pragma unroll
    for (int mi = 0; mi < 4; ++mi)
      af[mi] = frag_a(hp + (size_t)(mi * 16) * NC + k0, NC);
    #pragma unroll
    for (int ni = 0; ni < 2; ++ni) {
      const int nf   = wave * 2 + ni;
      const int nrow = (nf >> 2) * NC + u0 + (nf & 3) * 16;   // row in Whh [4096x1024]
      v16h bfr = frag_b(Wh + (size_t)nrow * NC + k0, NC);
      #pragma unroll
      for (int mi = 0; mi < 4; ++mi)
        acc[mi][ni] = __builtin_amdgcn_wmma_f32_16x16x32_f16(
            false, af[mi], false, bfr, (short)0, acc[mi][ni], false, false);
    }
  }

  // stage h@Whh^T pre-activations to LDS
  #pragma unroll
  for (int mi = 0; mi < 4; ++mi)
    #pragma unroll
    for (int ni = 0; ni < 2; ++ni) {
      const int nf = wave * 2 + ni;
      const int lc = (nf >> 2) * 64 + (nf & 3) * 16 + (lane & 15);
      const int mb = mi * 16 + (lane >> 4) * 8;
      #pragma unroll
      for (int r = 0; r < 8; ++r) gs[(mb + r) * 256 + lc] = acc[mi][ni][r];
    }
  __syncthreads();

  // pointwise gate math: 64x64 (b, uloc) pairs
  for (int idx = tid; idx < 64 * 64; idx += 256) {
    const int b  = idx >> 6;
    const int ul = idx & 63;
    const int u  = u0 + ul;
    const size_t xrow = ((size_t)b * NT + t) * NG;
    const float gi = gs[b * 256 +   0 + ul] + (float)xg[xrow +        u];
    const float gf = gs[b * 256 +  64 + ul] + (float)xg[xrow + NC   + u];
    const float gg = gs[b * 256 + 128 + ul] + (float)xg[xrow + 2*NC + u];
    const float go = gs[b * 256 + 192 + ul] + (float)xg[xrow + 3*NC + u];
    const float si = 1.f / (1.f + __expf(-gi));
    const float sf = 1.f / (1.f + __expf(-gf));
    const float so = 1.f / (1.f + __expf(-go));
    const size_t ci = (size_t)b * NC + u;
    const float cn = sf * cc[ci] + si * tanhf(gg);
    cc[ci] = cn;
    const float h = so * tanhf(cn);
    hn[ci] = (_Float16)h;
    out[((size_t)b * NT + t) * (2 * NC) + dir * NC + u] = h;
  }
}

// ---------------- host launcher ----------------
extern "C" void kernel_launch(void* const* d_in, const int* in_sizes, int n_in,
                              void* d_out, int out_size, void* d_ws, size_t ws_size,
                              hipStream_t stream) {
  (void)in_sizes; (void)n_in; (void)out_size; (void)ws_size;
  const float* x    = (const float*)d_in[0];
  const float* Wihf = (const float*)d_in[2];
  const float* Whhf = (const float*)d_in[3];
  const float* bf   = (const float*)d_in[4];
  const float* Wihb = (const float*)d_in[5];
  const float* Whhb = (const float*)d_in[6];
  const float* bb   = (const float*)d_in[7];
  const float* gx   = (const float*)d_in[8];
  const float* bx   = (const float*)d_in[9];
  const float* gH   = (const float*)d_in[10];
  const float* bH   = (const float*)d_in[11];
  float* out = (float*)d_out;

  char* ws = (char*)d_ws;
  size_t off = 0;
  auto take = [&](size_t bytes) {
    char* p = ws + off;
    off = (off + bytes + 255) & ~(size_t)255;
    return (void*)p;
  };
  _Float16* xn     = (_Float16*)take((size_t)NBT * NC * 2);   //  67 MB
  _Float16* wihf16 = (_Float16*)take((size_t)NG * NC * 2);    // 8.4 MB
  _Float16* whhf16 = (_Float16*)take((size_t)NG * NC * 2);
  _Float16* wihb16 = (_Float16*)take((size_t)NG * NC * 2);
  _Float16* whhb16 = (_Float16*)take((size_t)NG * NC * 2);
  _Float16* xgf    = (_Float16*)take((size_t)NBT * NG * 2);   // 268 MB
  _Float16* xgb    = (_Float16*)take((size_t)NBT * NG * 2);   // 268 MB
  _Float16* hA     = (_Float16*)take((size_t)2 * NB * NC * 2);
  _Float16* hB     = (_Float16*)take((size_t)2 * NB * NC * 2);
  float*    cbuf   = (float*)   take((size_t)2 * NB * NC * 4);

  const int wn = NG * NC;  // 4M elements per weight matrix
  f32_to_f16_kernel<<<wn / 256, 256, 0, stream>>>(Wihf, wihf16, wn);
  f32_to_f16_kernel<<<wn / 256, 256, 0, stream>>>(Whhf, whhf16, wn);
  f32_to_f16_kernel<<<wn / 256, 256, 0, stream>>>(Wihb, wihb16, wn);
  f32_to_f16_kernel<<<wn / 256, 256, 0, stream>>>(Whhb, whhb16, wn);

  ln_x_kernel<<<(int)NBT, 256, 0, stream>>>(x, gx, bx, xn);

  gemm_xproj_kernel<<<dim3(NG / 128, (int)(NBT / 128), 2), 256, 0, stream>>>(
      xn, wihf16, wihb16, bf, bb, xgf, xgb);

  init_state_kernel<<<512, 256, 0, stream>>>(hA, hB, cbuf);

  for (int s = 0; s < NT; ++s) {
    const _Float16* hp = (s & 1) ? hB : hA;
    _Float16*       hn = (s & 1) ? hA : hB;
    lstm_step_kernel<<<dim3(NC / 64, 1, 2), 256, 0, stream>>>(
        xgf, xgb, whhf16, whhb16, hp, hn, cbuf, out, s);
  }

  ln_out_kernel<<<(int)NBT, 256, 0, stream>>>(gH, bH, out);
}
